// SynthesisBlock_22325240005071
// MI455X (gfx1250) — compile-verified, matrix operand order checked by reading, not run on hardware
//
#include <hip/hip_runtime.h>

typedef __attribute__((ext_vector_type(16))) __bf16       v16bf;
typedef __attribute__((ext_vector_type(8)))  float        v8f;
typedef __attribute__((ext_vector_type(4)))  unsigned int u32x4;

#define CIN   512
#define COUT  512
#define WDIM  512
#define NB    16
#define HIN   32
#define HOUT  64
#define HP    66                     // padded spatial (1-pixel zero halo)
#define NPIX  (HOUT * HOUT)          // 4096
#define WPACK_ELEMS ((size_t)9 * 16 * 32 * 32 * 16)  // taps * icchunks * otiles * lanes * halves
#define XMOD_ELEMS  ((size_t)NB * HP * HP * CIN)     // padded bf16 activations

__device__ __forceinline__ unsigned short f2bf(float f) {
  unsigned int u = __float_as_uint(f);
  u += 0x7FFFu + ((u >> 16) & 1u);   // round-to-nearest-even
  return (unsigned short)(u >> 16);
}

// ---------------- zero the padded activation buffer (halo stays 0) --------------
__global__ void zero_kernel(u32x4* __restrict__ p) {
  size_t i = (size_t)blockIdx.x * 256 + threadIdx.x;
  p[i] = (u32x4){0, 0, 0, 0};
}

// ---------------- styles: out[b,o] = (w_lat[b] . W[o]) / sqrt(512) + bias[o] ----
__global__ void style_kernel(const float* __restrict__ wlat, const float* __restrict__ W,
                             const float* __restrict__ bias, float* __restrict__ out) {
  int idx = blockIdx.x * 256 + threadIdx.x;     // 16*512
  int b = idx >> 9, o = idx & 511;
  const float* wl = wlat + (size_t)b * WDIM;
  const float* wr = W + (size_t)o * WDIM;
  float acc = 0.f;
#pragma unroll 8
  for (int k = 0; k < WDIM; ++k) acc += wl[k] * wr[k];
  out[idx] = acc * 0.04419417382415922f + bias[o];   // 1/sqrt(512)
}

// ---------------- S[o,i] = wg^2 * sum_k w[o,i,k]^2 ------------------------------
__global__ void sumsq_kernel(const float* __restrict__ w, float* __restrict__ S) {
  int idx = blockIdx.x * 256 + threadIdx.x;     // 512*512
  const float* p = w + (size_t)idx * 9;
  float s = 0.f;
#pragma unroll
  for (int k = 0; k < 9; ++k) s += p[k] * p[k];
  S[idx] = s * (1.0f / 4608.0f);                // wg^2 = 1/(Cin*9)
}

// ---------------- demod[b,o] = rsqrt(sum_i style^2 * S + 1e-8) ------------------
__global__ void demod_kernel(const float* __restrict__ style, const float* __restrict__ S,
                             float* __restrict__ dm) {
  int idx = blockIdx.x * 256 + threadIdx.x;     // 16*512
  int b = idx >> 9, o = idx & 511;
  const float* st = style + (size_t)b * CIN;
  const float* sr = S + (size_t)o * CIN;
  float acc = 1e-8f;
#pragma unroll 8
  for (int i = 0; i < CIN; ++i) { float s = st[i]; acc += s * s * sr[i]; }
  dm[idx] = rsqrtf(acc);
}

// ---------------- pack weights*wg into per-lane bf16 WMMA A-fragments -----------
// layout: [t 9][ic 16][otile 32][lane 32][half 16]; ISA 16-bit A 16x32 layout.
__global__ void pack_kernel(const float* __restrict__ w, unsigned short* __restrict__ wp) {
  int idx = blockIdx.x * 256 + threadIdx.x;     // 2,359,296
  int j    = idx & 15;
  int lane = (idx >> 4) & 31;
  int ot   = (idx >> 9) & 31;
  int ic   = (idx >> 14) & 15;
  int t    = idx >> 18;                          // 0..8
  int m  = lane & 15;
  int kb = (lane & 16) ? 8 : 0;
  int K  = kb + ((j < 8) ? j : (j + 8));         // {0..7,16..23} or {8..15,24..31}
  int o = ot * 16 + m;
  int i = ic * 32 + K;
  float v = w[((size_t)o * CIN + i) * 9 + t] * 0.014731391274719739f; // 1/sqrt(4608)
  wp[idx] = f2bf(v);
}

// ---------------- bilinear coefs (half-pixel, edge clamp, scale 2) --------------
__device__ __forceinline__ void lincoef(int j, int& a, int& b, float& wa, float& wb) {
  int s0; float f;
  if (j & 1) { s0 = j >> 1; f = 0.25f; } else { s0 = (j >> 1) - 1; f = 0.75f; }
  a = s0 < 0 ? 0 : s0;
  b = (s0 + 1 > HIN - 1) ? (HIN - 1) : (s0 + 1);
  wa = 1.f - f; wb = f;
}

// ------- upsample 2x + modulate; write channels-last bf16 into padded buf -------
__global__ void upsample_mod_kernel(const float* __restrict__ x, const float* __restrict__ style,
                                    unsigned short* __restrict__ xmod) {
  __shared__ unsigned short lds[64 * 32];
  int ic = blockIdx.x, y = blockIdx.y, b = blockIdx.z;
  int ya, yb; float wya, wyb;
  lincoef(y, ya, yb, wya, wyb);
  int i0 = ic * 32;
#pragma unroll
  for (int r = 0; r < 8; ++r) {
    int idx = threadIdx.x + 256 * r;
    int i = idx >> 6, xcol = idx & 63;
    int xa, xb; float wxa, wxb;
    lincoef(xcol, xa, xb, wxa, wxb);
    const float* base = x + (size_t)(b * CIN + i0 + i) * (HIN * HIN);
    float v = wya * (wxa * base[ya * HIN + xa] + wxb * base[ya * HIN + xb]) +
              wyb * (wxa * base[yb * HIN + xa] + wxb * base[yb * HIN + xb]);
    v *= style[b * CIN + i0 + i];
    lds[xcol * 32 + i] = f2bf(v);
  }
  __syncthreads();
  unsigned short* dst = xmod + ((size_t)(b * HP + y + 1) * HP + 1) * CIN + i0;
#pragma unroll
  for (int r = 0; r < 8; ++r) {
    int idx = threadIdx.x + 256 * r;
    int i = idx & 31, xcol = idx >> 5;
    dst[(size_t)xcol * CIN + i] = lds[xcol * 32 + i];
  }
}

// ---------------- implicit-GEMM conv 3x3 via bf16 WMMA, fused epilogue ----------
// block = 8 waves -> 64 outch x 128 pixels (two image rows).
// wave = 2x2 C tiles (32 outch x 32 pix): 2 A frags x 2 B frags -> 2 loads/WMMA.
// K = 9 taps x 16 chunks of 32 ch; zero halo => branchless inner loop.
__global__ void __launch_bounds__(256)
conv_wmma_kernel(const unsigned short* __restrict__ xmod,
                 const unsigned short* __restrict__ wpack,
                 const float* __restrict__ dm,
                 const float* __restrict__ noise,
                 const float* __restrict__ nwt,
                 float* __restrict__ out) {
  __shared__ float ldsC[64 * 129];
  int ot = blockIdx.x;                    // 0..7 (64-outch tile)
  int y0 = blockIdx.y * 2;                // first of two image rows
  int b  = blockIdx.z;

  int tid  = threadIdx.x;
  int wave = tid >> 5, lane = tid & 31;
  int oSub = wave & 1;                    // 2 o-groups of 32 outch
  int pSub = wave >> 1;                   // 4 pixel groups of 32 px
  int rowSel = pSub >> 1;                 // which of the two rows
  int xbase  = (pSub & 1) * 32;           // x of first B fragment
  int otA  = ot * 4 + oSub * 2;           // first of two 16-outch tiles

  int n    = lane & 15;
  int koff = (lane & 16) ? 8 : 0;

  union FragU { u32x4 q[2]; v16bf v; };
  v8f c00 = {}, c01 = {}, c10 = {}, c11 = {};

  // padded coords: tap (dy,dx) reads pixel (y+dy, x+dx)
  const unsigned short* xb =
      xmod + ((size_t)(b * HP + y0 + rowSel) * HP + xbase + n) * CIN + koff;

  for (int t = 0; t < 9; ++t) {
    int dy = t / 3, dx = t - dy * 3;
    const unsigned short* bp = xb + (size_t)(dy * HP + dx) * CIN;
    const unsigned short* ap = wpack + (size_t)((t * 16 * 32 + otA) * 32 + lane) * 16;
#pragma unroll 4
    for (int ic = 0; ic < 16; ++ic) {
      FragU A0, A1, B0, B1;
      const u32x4* a4 = (const u32x4*)(ap + (size_t)ic * (32 * 32 * 16));
      A0.q[0] = a4[0];                     // otile otA
      A0.q[1] = a4[1];
      A1.q[0] = a4[64];                    // otile otA+1 (+512 halves)
      A1.q[1] = a4[65];
      const unsigned short* bc = bp + ic * 32;
      B0.q[0] = *(const u32x4*)(bc);                  // px xbase+n,    ch koff..+7
      B0.q[1] = *(const u32x4*)(bc + 16);             //                ch koff+16..+23
      B1.q[0] = *(const u32x4*)(bc + 16 * CIN);       // px xbase+16+n
      B1.q[1] = *(const u32x4*)(bc + 16 * CIN + 16);
      c00 = __builtin_amdgcn_wmma_f32_16x16x32_bf16(false, A0.v, false, B0.v,
                                                    (short)0, c00, false, false);
      c10 = __builtin_amdgcn_wmma_f32_16x16x32_bf16(false, A1.v, false, B0.v,
                                                    (short)0, c10, false, false);
      c01 = __builtin_amdgcn_wmma_f32_16x16x32_bf16(false, A0.v, false, B1.v,
                                                    (short)0, c01, false, false);
      c11 = __builtin_amdgcn_wmma_f32_16x16x32_bf16(false, A1.v, false, B1.v,
                                                    (short)0, c11, false, false);
    }
  }

  // C fragments -> LDS (ISA 32-bit C/D 16x16 layout), then coalesced NCHW epilogue
  {
    int mb = (lane & 16) ? 8 : 0;
    int p0 = rowSel * 64 + xbase + n;     // pixel index within 128-px block tile
#pragma unroll
    for (int r = 0; r < 8; ++r) {
      int oa = oSub * 32 + mb + r;
      ldsC[oa * 129 + p0]             = c00[r];
      ldsC[(oa + 16) * 129 + p0]      = c10[r];
      ldsC[oa * 129 + p0 + 16]        = c01[r];
      ldsC[(oa + 16) * 129 + p0 + 16] = c11[r];
    }
  }
  __syncthreads();
  int o0 = ot * 64;
#pragma unroll
  for (int r = 0; r < 32; ++r) {
    int idx = tid + 256 * r;
    int o = idx >> 7, p = idx & 127;      // p: row = p>>6, x = p&63
    int row = y0 + (p >> 6), xc = p & 63;
    float v = ldsC[o * 129 + p] * dm[b * COUT + o0 + o];
    v += nwt[o0 + o] * noise[(size_t)(b * HOUT + row) * HOUT + xc];
    v = (v > 0.f) ? v : 0.2f * v;         // leaky_relu 0.2
    out[(size_t)(b * COUT + o0 + o) * NPIX + row * HOUT + xc] = v;
  }
}

// ---------------- per (b,o) instance-norm stats ---------------------------------
__global__ void stats_kernel(const float* __restrict__ h, float* __restrict__ mean,
                             float* __restrict__ rstd) {
  __shared__ float ss[256], sq[256];
  int bo = blockIdx.x;
  const float* p = h + (size_t)bo * NPIX;
  float s = 0.f, q = 0.f;
  for (int k = threadIdx.x; k < NPIX; k += 256) { float v = p[k]; s += v; q += v * v; }
  ss[threadIdx.x] = s; sq[threadIdx.x] = q;
  __syncthreads();
  for (int st = 128; st > 0; st >>= 1) {
    if (threadIdx.x < st) {
      ss[threadIdx.x] += ss[threadIdx.x + st];
      sq[threadIdx.x] += sq[threadIdx.x + st];
    }
    __syncthreads();
  }
  if (threadIdx.x == 0) {
    float m = ss[0] * (1.f / NPIX);
    float var = sq[0] * (1.f / NPIX) - m * m;
    mean[bo] = m;
    rstd[bo] = rsqrtf(var + 1e-5f);
  }
}

// --- AdaIN apply + modulate for next conv; write channels-last bf16 (padded) ----
__global__ void adain_mod_kernel(const float* __restrict__ h, const float* __restrict__ mean,
                                 const float* __restrict__ rstd, const float* __restrict__ as,
                                 const float* __restrict__ ah, const float* __restrict__ stn,
                                 unsigned short* __restrict__ xmod) {
  __shared__ unsigned short lds[64 * 32];
  int oc = blockIdx.x, y = blockIdx.y, b = blockIdx.z;
  int o0 = oc * 32;
#pragma unroll
  for (int r = 0; r < 8; ++r) {
    int idx = threadIdx.x + 256 * r;
    int o = idx >> 6, xcol = idx & 63;
    int bo = b * COUT + o0 + o;
    float v = h[(size_t)bo * NPIX + y * HOUT + xcol];
    v = (v - mean[bo]) * rstd[bo];
    v = as[bo] * v + ah[bo];
    v *= stn[bo];
    lds[xcol * 32 + o] = f2bf(v);
  }
  __syncthreads();
  unsigned short* dst = xmod + ((size_t)(b * HP + y + 1) * HP + 1) * CIN + o0;
#pragma unroll
  for (int r = 0; r < 8; ++r) {
    int idx = threadIdx.x + 256 * r;
    int o = idx & 31, xcol = idx >> 5;
    dst[(size_t)xcol * CIN + o] = lds[xcol * 32 + o];
  }
}

// ---------------- final AdaIN, in place on d_out --------------------------------
__global__ void final_adain_kernel(float* __restrict__ h, const float* __restrict__ mean,
                                   const float* __restrict__ rstd, const float* __restrict__ as,
                                   const float* __restrict__ ah) {
  size_t idx = (size_t)blockIdx.x * 256 + threadIdx.x;
  int bo = (int)(idx >> 12);
  float v = h[idx];
  v = (v - mean[bo]) * rstd[bo];
  h[idx] = as[bo] * v + ah[bo];
}

extern "C" void kernel_launch(void* const* d_in, const int* in_sizes, int n_in,
                              void* d_out, int out_size, void* d_ws, size_t ws_size,
                              hipStream_t stream) {
  (void)in_sizes; (void)n_in; (void)out_size; (void)ws_size;
  const float* x       = (const float*)d_in[0];
  const float* w0      = (const float*)d_in[1];
  const float* w1      = (const float*)d_in[2];
  const float* noise0  = (const float*)d_in[3];
  const float* noise1  = (const float*)d_in[4];
  const float* mod_w0  = (const float*)d_in[5];
  const float* mod_b0  = (const float*)d_in[6];
  const float* conv_w0 = (const float*)d_in[7];
  const float* nwt0    = (const float*)d_in[8];
  const float* a0sw    = (const float*)d_in[9];
  const float* a0sb    = (const float*)d_in[10];
  const float* a0hw    = (const float*)d_in[11];
  const float* a0hb    = (const float*)d_in[12];
  const float* mod_w1  = (const float*)d_in[13];
  const float* mod_b1  = (const float*)d_in[14];
  const float* conv_w1 = (const float*)d_in[15];
  const float* nwt1    = (const float*)d_in[16];
  const float* a1sw    = (const float*)d_in[17];
  const float* a1sb    = (const float*)d_in[18];
  const float* a1hw    = (const float*)d_in[19];
  const float* a1hb    = (const float*)d_in[20];
  float* out = (float*)d_out;

  char* ws = (char*)d_ws;
  size_t off = 0;
  auto carve = [&](size_t bytes) -> char* {
    char* p = ws + off;
    off += (bytes + 255) & ~(size_t)255;
    return p;
  };
  unsigned short* xmod = (unsigned short*)carve(XMOD_ELEMS * 2);   // 71 MB padded
  unsigned short* wp0  = (unsigned short*)carve(WPACK_ELEMS * 2);  // 4.7 MB
  unsigned short* wp1  = (unsigned short*)carve(WPACK_ELEMS * 2);
  float* S0  = (float*)carve((size_t)COUT * CIN * 4);              // 1 MB
  float* S1  = (float*)carve((size_t)COUT * CIN * 4);
  float* st0 = (float*)carve((size_t)NB * CIN * 4);
  float* st1 = (float*)carve((size_t)NB * CIN * 4);
  float* a0s = (float*)carve((size_t)NB * COUT * 4);
  float* a0h = (float*)carve((size_t)NB * COUT * 4);
  float* a1s = (float*)carve((size_t)NB * COUT * 4);
  float* a1h = (float*)carve((size_t)NB * COUT * 4);
  float* dm0 = (float*)carve((size_t)NB * COUT * 4);
  float* dm1 = (float*)carve((size_t)NB * COUT * 4);
  float* mn  = (float*)carve((size_t)NB * COUT * 4);
  float* rs  = (float*)carve((size_t)NB * COUT * 4);

  // zero padded activation buffer (halo must be 0 every call)
  zero_kernel<<<(unsigned)(XMOD_ELEMS * 2 / 16 / 256), 256, 0, stream>>>((u32x4*)xmod);

  style_kernel<<<32, 256, 0, stream>>>(w0, mod_w0, mod_b0, st0);
  style_kernel<<<32, 256, 0, stream>>>(w1, mod_w1, mod_b1, st1);
  style_kernel<<<32, 256, 0, stream>>>(w0, a0sw, a0sb, a0s);
  style_kernel<<<32, 256, 0, stream>>>(w0, a0hw, a0hb, a0h);
  style_kernel<<<32, 256, 0, stream>>>(w1, a1sw, a1sb, a1s);
  style_kernel<<<32, 256, 0, stream>>>(w1, a1hw, a1hb, a1h);
  sumsq_kernel<<<1024, 256, 0, stream>>>(conv_w0, S0);
  sumsq_kernel<<<1024, 256, 0, stream>>>(conv_w1, S1);
  demod_kernel<<<32, 256, 0, stream>>>(st0, S0, dm0);
  demod_kernel<<<32, 256, 0, stream>>>(st1, S1, dm1);
  pack_kernel<<<9216, 256, 0, stream>>>(conv_w0, wp0);
  pack_kernel<<<9216, 256, 0, stream>>>(conv_w1, wp1);

  upsample_mod_kernel<<<dim3(16, 64, 16), 256, 0, stream>>>(x, st0, xmod);
  conv_wmma_kernel<<<dim3(8, 32, 16), 256, 0, stream>>>(xmod, wp0, dm0, noise0, nwt0, out);
  stats_kernel<<<NB * COUT, 256, 0, stream>>>(out, mn, rs);
  adain_mod_kernel<<<dim3(16, 64, 16), 256, 0, stream>>>(out, mn, rs, a0s, a0h, st1, xmod);
  conv_wmma_kernel<<<dim3(8, 32, 16), 256, 0, stream>>>(xmod, wp1, dm1, noise1, nwt1, out);
  stats_kernel<<<NB * COUT, 256, 0, stream>>>(out, mn, rs);
  final_adain_kernel<<<131072, 256, 0, stream>>>(out, mn, rs, a1s, a1h);
}
